// Head_80461917323908
// MI455X (gfx1250) — matrix operand order repaired and checked by reading, not compiled, not run on hardware
//
#include <hip/hip_runtime.h>
#include <hip/hip_bf16.h>

// HEAD attention: x[32,1024,1024] f32, pad[32,1024] i32, Wk/Wq/Wv[1024,64] f32
// out[32,1024,64] f32.
// Workspace: Qh, Kh (f16 [B*T,64]), Vt (f16 [B,64,T]), Wt (f16 [3*64,1024]) = ~12.4 MB.

typedef _Float16 h16v __attribute__((ext_vector_type(16)));
typedef _Float16 h8v  __attribute__((ext_vector_type(8)));
typedef float    f8v  __attribute__((ext_vector_type(8)));
typedef float    f4v  __attribute__((ext_vector_type(4)));

#define T_SEQ   1024
#define C_DIM   1024
#define H_DIM   64
#define XF_STR  76   // floats; 304B rows: 16B aligned, 12-bank shift => conflict-free
#define WT_STR  72   // halfs;  144B rows: 16B aligned, 36-bank shift => conflict-free
#define PB_STR  48   // halfs;   96B rows: 16B aligned

__device__ __forceinline__ f8v wmma16(h16v a, h16v b, f8v c) {
  // emits v_wmma_f32_16x16x32_f16
  return __builtin_amdgcn_wmma_f32_16x16x32_f16(false, a, false, b, (short)0, c,
                                                false, false);
}

// ---- CDNA5 async global->LDS copy (ASYNCcnt-tracked), 16B per lane
__device__ __forceinline__ void async_copy16(unsigned lds_off, const void* gptr) {
  asm volatile("global_load_async_to_lds_b128 %0, %1, off"
               :: "v"(lds_off), "v"((unsigned long long)gptr) : "memory");
}
__device__ __forceinline__ void async_wait_le10() {
  asm volatile("s_wait_asynccnt 0xa" ::: "memory");
}
__device__ __forceinline__ void async_wait_0() {
  asm volatile("s_wait_asynccnt 0x0" ::: "memory");
}

// B fragment (32x16 KxN f16): lane holds 16 contiguous K-halves of its column.
__device__ __forceinline__ h16v load_b16(const _Float16* p) {
  h8v lo = *(const h8v*)p;
  h8v hi = *(const h8v*)(p + 8);
  return __builtin_shufflevector(lo, hi, 0,1,2,3,4,5,6,7,8,9,10,11,12,13,14,15);
}

// A fragment (16x32 MxK f16) from f16 source; rowp = row start for this lane.
__device__ __forceinline__ h16v load_a16(const _Float16* rowp, int kk, int hf) {
  const _Float16* p = rowp + kk + hf * 16;
  h8v lo = *(const h8v*)p;
  h8v hi = *(const h8v*)(p + 8);
  return __builtin_shufflevector(lo, hi, 0,1,2,3,4,5,6,7,8,9,10,11,12,13,14,15);
}

// A fragment from f32 LDS tile with on-the-fly f32->f16 (co-executes with WMMA)
__device__ __forceinline__ h16v load_a16_f32(const float* rowp, int kk, int hf) {
  const float* p = rowp + kk + hf * 16;
  f4v a = *(const f4v*)(p + 0);
  f4v b = *(const f4v*)(p + 4);
  f4v c = *(const f4v*)(p + 8);
  f4v d = *(const f4v*)(p + 12);
  h16v r = {(_Float16)a[0], (_Float16)a[1], (_Float16)a[2], (_Float16)a[3],
            (_Float16)b[0], (_Float16)b[1], (_Float16)b[2], (_Float16)b[3],
            (_Float16)c[0], (_Float16)c[1], (_Float16)c[2], (_Float16)c[3],
            (_Float16)d[0], (_Float16)d[1], (_Float16)d[2], (_Float16)d[3]};
  return r;
}

// ---------------------------------------------- stage 0: W -> f16, transposed
// Wt[(m*64 + n)][k=0..1023], m: 0=Wk, 1=Wq, 2=Wv
__global__ __launch_bounds__(256) void wtrans_kernel(
    const float* __restrict__ Wk, const float* __restrict__ Wq,
    const float* __restrict__ Wv, _Float16* __restrict__ Wt) {
  int idx = blockIdx.x * 256 + threadIdx.x;   // 0..24575
  int m   = idx >> 13;
  int rem = idx & 8191;
  int n   = rem & 63;
  int k0  = (rem >> 6) << 3;
  const float* W = (m == 0) ? Wk : (m == 1) ? Wq : Wv;
  h8v p;
#pragma unroll
  for (int i = 0; i < 8; ++i)
    p[i] = (_Float16)W[(size_t)(k0 + i) * H_DIM + n];
  *(h8v*)(Wt + (((size_t)(m * 64 + n)) << 10) + k0) = p;
}

// ---------------------------------------------------------------- stage 1: QKV
__global__ __launch_bounds__(256) void qkv_proj_kernel(
    const float* __restrict__ x, const _Float16* __restrict__ Wt,
    _Float16* __restrict__ Qh, _Float16* __restrict__ Kh,
    _Float16* __restrict__ Vt) {
  __shared__ float    xs[2][64 * XF_STR];       // raw f32 x tiles (async-staged)
  __shared__ _Float16 wls[2][3 * 64 * WT_STR];  // f16 transposed W (async-staged)

  const int tid  = threadIdx.x;
  const int lane = tid & 31;
  const int ln   = lane & 15;
  const int hf   = lane >> 4;
  const int wave = tid >> 5;
  const int rowbase = blockIdx.x * 64;          // flattened (b*T + t) row base
  const int wrow = (wave >> 1) * 16;
  const int wcol = (wave & 1) * 32;

  f8v acc[3][2];
#pragma unroll
  for (int m = 0; m < 3; ++m)
#pragma unroll
    for (int c = 0; c < 2; ++c)
#pragma unroll
      for (int v = 0; v < 8; ++v) acc[m][c][v] = 0.0f;

  // async stage of one 64-wide k-chunk: exactly 10 b128 copies per thread
  auto issue_chunk = [&](int kc, int buf) {
#pragma unroll
    for (int i = 0; i < 4; ++i) {               // x: 64 rows x 256B
      int id = tid + i * 256;                   // 0..1023
      int r  = id >> 4;
      int c4 = (id & 15) << 2;
      const float* gp = x + (size_t)(rowbase + r) * C_DIM + kc + c4;
      unsigned lo = (unsigned)(uintptr_t)(&xs[buf][r * XF_STR + c4]);
      async_copy16(lo, gp);
    }
#pragma unroll
    for (int j = 0; j < 6; ++j) {               // Wt: 3 x 64 cols x 128B
      int id  = tid + j * 256;                  // 0..1535
      int mat = id >> 9;
      int rem = id & 511;
      int n   = rem >> 3;
      int h0  = (rem & 7) << 3;
      const _Float16* gp = Wt + (((size_t)(mat * 64 + n)) << 10) + kc + h0;
      unsigned lo =
          (unsigned)(uintptr_t)(&wls[buf][(mat * 64 + n) * WT_STR + h0]);
      async_copy16(lo, gp);
    }
  };

  issue_chunk(0, 0);                            // prologue
  for (int ci = 0; ci < 16; ++ci) {
    const int buf = ci & 1;
    const bool has_next = (ci + 1) < 16;
    if (has_next) issue_chunk((ci + 1) * 64, buf ^ 1);   // overlap with compute
    if (has_next) async_wait_le10();            // chunk ci landed; next in flight
    else          async_wait_0();
    __syncthreads();

    const float* arow = &xs[buf][(wrow + ln) * XF_STR];
#pragma unroll
    for (int kk = 0; kk < 64; kk += 32) {
      h16v afrag = load_a16_f32(arow, kk, hf);
#pragma unroll
      for (int m = 0; m < 3; ++m)
#pragma unroll
        for (int ct = 0; ct < 2; ++ct) {
          const _Float16* bp =
              &wls[buf][(m * 64 + wcol + ct * 16 + ln) * WT_STR + kk + hf * 16];
          acc[m][ct] = wmma16(afrag, load_b16(bp), acc[m][ct]);
        }
    }
    __syncthreads();                            // buf reusable for chunk ci+2
  }

  // ---- store: K->Kh, Q->Qh row-major f16; V->Vt transposed (packed 16B)
  const int mb = hf * 8;
  const int bb = rowbase >> 10;
  const int tb = (rowbase & (T_SEQ - 1)) + wrow + mb;
#pragma unroll
  for (int ct = 0; ct < 2; ++ct) {
    int col = wcol + ct * 16 + ln;
    f8v ka = acc[0][ct], qa = acc[1][ct], va = acc[2][ct];
#pragma unroll
    for (int v = 0; v < 8; ++v) {
      size_t r = (size_t)(rowbase + wrow + mb + v);
      Kh[r * H_DIM + col] = (_Float16)ka[v];
      Qh[r * H_DIM + col] = (_Float16)qa[v];
    }
    h8v pk = {(_Float16)va[0], (_Float16)va[1], (_Float16)va[2], (_Float16)va[3],
              (_Float16)va[4], (_Float16)va[5], (_Float16)va[6], (_Float16)va[7]};
    *(h8v*)(Vt + (((size_t)bb * H_DIM + col) << 10) + tb) = pk;
  }
}

// -------------------------------------------------------- stage 2: attention
__global__ __launch_bounds__(256) void attn_kernel(
    const _Float16* __restrict__ Qh, const _Float16* __restrict__ Kh,
    const _Float16* __restrict__ Vt, const int* __restrict__ pad,
    float* __restrict__ out) {
  __shared__ _Float16 pbuf[8][16 * PB_STR];   // wave-private P staging

  const int tid  = threadIdx.x;
  const int lane = tid & 31;
  const int ln   = lane & 15;
  const int hf   = lane >> 4;
  const int wave = tid >> 5;
  const int b    = blockIdx.y;
  const int i0   = blockIdx.x * 128 + wave * 16;   // wave's 16-row strip
  const int mb   = hf * 8;
  const float scale = 0.125f;                      // 1/sqrt(64)

  const _Float16* qrow = Qh + ((size_t)(b << 10) + i0 + ln) * H_DIM;
  h16v aq0 = load_a16(qrow, 0, hf);
  h16v aq1 = load_a16(qrow, 32, hf);

  bool padv[8];
#pragma unroll
  for (int v = 0; v < 8; ++v)
    padv[v] = (pad[(b << 10) + i0 + mb + v] == 0);

  float mrun[8], lrun[8];
  f8v accO[4];
#pragma unroll
  for (int v = 0; v < 8; ++v) { mrun[v] = -__builtin_inff(); lrun[v] = 0.0f; }
#pragma unroll
  for (int ht = 0; ht < 4; ++ht)
#pragma unroll
    for (int v = 0; v < 8; ++v) accO[ht][v] = 0.0f;

  const _Float16* kbase = Kh + ((size_t)(b << 10)) * H_DIM;
  const _Float16* vbase = Vt + ((size_t)b * H_DIM << 10);
  _Float16* pb = pbuf[wave];

  const int jend = i0 + 16;                   // causal: j <= i < i0+16
  for (int j0 = 0; j0 < jend; j0 += 32) {
    f8v s0, s1;
#pragma unroll
    for (int v = 0; v < 8; ++v) { s0[v] = 0.0f; s1[v] = 0.0f; }
    {
      const _Float16* k0 = kbase + (size_t)(j0 + ln) * H_DIM + hf * 16;
      const _Float16* k1 = kbase + (size_t)(j0 + 16 + ln) * H_DIM + hf * 16;
      s0 = wmma16(aq0, load_b16(k0), s0);
      s0 = wmma16(aq1, load_b16(k0 + 32), s0);
      s1 = wmma16(aq0, load_b16(k1), s1);
      s1 = wmma16(aq1, load_b16(k1 + 32), s1);
    }
    float sv0[8], sv1[8], rm[8];
#pragma unroll
    for (int v = 0; v < 8; ++v) {
      int i = i0 + mb + v;
      float a0 = s0[v] * scale, a1 = s1[v] * scale;
      if (padv[v]) { a0 = -1e9f; a1 = -1e9f; }
      if (j0 + ln > i)      a0 = -__builtin_inff();
      if (j0 + 16 + ln > i) a1 = -__builtin_inff();
      sv0[v] = a0; sv1[v] = a1;
      rm[v] = fmaxf(a0, a1);
    }
#pragma unroll
    for (int off = 1; off < 16; off <<= 1)
#pragma unroll
      for (int v = 0; v < 8; ++v)
        rm[v] = fmaxf(rm[v], __shfl_xor(rm[v], off, 32));

    float co[8], rs[8], p0[8], p1[8];
#pragma unroll
    for (int v = 0; v < 8; ++v) {
      float mn = fmaxf(mrun[v], rm[v]);
      co[v] = __expf(mrun[v] - mn);
      p0[v] = __expf(sv0[v] - mn);
      p1[v] = __expf(sv1[v] - mn);
      rs[v] = p0[v] + p1[v];
      mrun[v] = mn;
    }
#pragma unroll
    for (int off = 1; off < 16; off <<= 1)
#pragma unroll
      for (int v = 0; v < 8; ++v)
        rs[v] += __shfl_xor(rs[v], off, 32);
#pragma unroll
    for (int v = 0; v < 8; ++v) lrun[v] = lrun[v] * co[v] + rs[v];
#pragma unroll
    for (int ht = 0; ht < 4; ++ht)
#pragma unroll
      for (int v = 0; v < 8; ++v) accO[ht][v] *= co[v];

    // C-layout P -> LDS -> A-layout fragment (wave-private, no barrier)
#pragma unroll
    for (int v = 0; v < 8; ++v) {
      int mr = mb + v;
      pb[mr * PB_STR + ln]      = (_Float16)p0[v];
      pb[mr * PB_STR + 16 + ln] = (_Float16)p1[v];
    }
    asm volatile("s_wait_dscnt 0x0" ::: "memory");
    h16v ap = load_a16(pb + ln * PB_STR, 0, hf);

#pragma unroll
    for (int ht = 0; ht < 4; ++ht) {
      const _Float16* vp = vbase + (((size_t)(ht * 16 + ln)) << 10) + j0 + hf * 16;
      accO[ht] = wmma16(ap, load_b16(vp), accO[ht]);
    }
  }

  float inv[8];
#pragma unroll
  for (int v = 0; v < 8; ++v) inv[v] = 1.0f / lrun[v];
#pragma unroll
  for (int ht = 0; ht < 4; ++ht) {
    int h = ht * 16 + ln;
#pragma unroll
    for (int v = 0; v < 8; ++v) {
      size_t r = (size_t)(b << 10) + i0 + mb + v;
      out[r * H_DIM + h] = accO[ht][v] * inv[v];
    }
  }
}

extern "C" void kernel_launch(void* const* d_in, const int* in_sizes, int n_in,
                              void* d_out, int out_size, void* d_ws, size_t ws_size,
                              hipStream_t stream) {
  (void)in_sizes; (void)n_in; (void)out_size; (void)ws_size;
  const float* x   = (const float*)d_in[0];
  const int*   pad = (const int*)d_in[1];
  const float* Wk  = (const float*)d_in[2];
  const float* Wq  = (const float*)d_in[3];
  const float* Wv  = (const float*)d_in[4];
  float* out = (float*)d_out;

  // workspace: Qh | Kh | Vt (4 MB each) | Wt (384 KB)
  _Float16* Qh = (_Float16*)d_ws;
  _Float16* Kh = Qh + (size_t)32 * T_SEQ * H_DIM;
  _Float16* Vt = Kh + (size_t)32 * T_SEQ * H_DIM;
  _Float16* Wt = Vt + (size_t)32 * T_SEQ * H_DIM;

  wtrans_kernel<<<96, 256, 0, stream>>>(Wk, Wq, Wv, Wt);
  qkv_proj_kernel<<<512, 256, 0, stream>>>(x, Wt, Qh, Kh, Vt);
  attn_kernel<<<dim3(8, 32), 256, 0, stream>>>(Qh, Kh, Vt, pad, out);
}
